// ScaledDotProductAttention_13030930776119
// MI455X (gfx1250) — matrix operand order repaired and checked by reading, not compiled, not run on hardware
//
#include <hip/hip_runtime.h>

typedef __attribute__((ext_vector_type(16))) _Float16 v16h;
typedef __attribute__((ext_vector_type(8)))  _Float16 v8h;
typedef __attribute__((ext_vector_type(4)))  _Float16 v4h;
typedef __attribute__((ext_vector_type(8)))  float    v8f;
typedef __attribute__((ext_vector_type(4)))  float    v4f;

constexpr int B_ = 64, S_ = 2048, D_ = 128;
constexpr int QT = 16;               // q rows per workgroup
constexpr int SSTR = S_ + 8;         // LDS score-row stride (halves), padded
constexpr float INV_TEMPER = 0.08838834764831845f;  // 1/sqrt(128)

__device__ inline float wave_max(float x) {
  for (int off = 16; off > 0; off >>= 1) x = fmaxf(x, __shfl_xor(x, off, 32));
  return x;
}
__device__ inline float wave_sum(float x) {
  for (int off = 16; off > 0; off >>= 1) x += __shfl_xor(x, off, 32);
  return x;
}

// ---- prep: elementwise f32 -> f16 (for K) --------------------------------
__global__ __launch_bounds__(256)
void cvt_f32_to_f16(const float* __restrict__ in, _Float16* __restrict__ out,
                    int n4) {
  int i = blockIdx.x * blockDim.x + threadIdx.x;
  if (i < n4) {
    v4f t = ((const v4f*)in)[i];
    v4h o;
#pragma unroll
    for (int e = 0; e < 4; ++e) o[e] = (_Float16)t[e];
    ((v4h*)out)[i] = o;
  }
}

// ---- prep: V [B,S,D] f32 -> Vt [B,D,S] f16 (tiled transpose) -------------
__global__ __launch_bounds__(256)
void transpose_v(const float* __restrict__ v, _Float16* __restrict__ vt) {
  __shared__ _Float16 tile[32][33];
  const int b = blockIdx.z;
  const int s0 = blockIdx.y * 32, d0 = blockIdx.x * 32;
  const int tx = threadIdx.x, ty = threadIdx.y;
#pragma unroll
  for (int i = 0; i < 4; ++i) {
    int r = ty + i * 8;
    tile[r][tx] = (_Float16)v[((size_t)b * S_ + s0 + r) * D_ + d0 + tx];
  }
  __syncthreads();
#pragma unroll
  for (int i = 0; i < 4; ++i) {
    int r = ty + i * 8;
    vt[((size_t)b * D_ + d0 + r) * S_ + s0 + tx] = tile[tx][r];
  }
}

// ---- fused attention: scores -> softmax (+att out) -> out = att*V --------
template <bool PREP>
__global__ __launch_bounds__(256)
void attn_kernel(const float* __restrict__ q, const float* __restrict__ k,
                 const float* __restrict__ v, const _Float16* __restrict__ kh,
                 const _Float16* __restrict__ vt, float* __restrict__ outp,
                 float* __restrict__ att) {
  extern __shared__ char smem_raw[];
  _Float16* sc  = (_Float16*)smem_raw;                       // [QT][SSTR]
  float*    rs16 = (float*)(smem_raw + QT * SSTR * sizeof(_Float16)); // [QT]

  const int b    = blockIdx.x >> 7;       // / (S_/QT)
  const int q0   = (blockIdx.x & 127) * QT;
  const int wv   = __builtin_amdgcn_readfirstlane(threadIdx.x >> 5); // scalar
  const int lane = threadIdx.x & 31;
  const int m    = lane & 15;
  const int h    = lane >> 4;
  const int khalf = h * 8;

  // -- Q tile as four 16x32 f16 A-fragments (contiguous k = 16h + s) --
  v16h afrag[4];
  {
    const float* qrow = q + ((size_t)b * S_ + q0 + m) * D_;
#pragma unroll
    for (int f = 0; f < 4; ++f) {
      const int kb = f * 32;
      const v4f* p0 = (const v4f*)(qrow + kb + 16 * h);
      const v4f* p1 = (const v4f*)(qrow + kb + 16 * h + 8);
      v4f t0 = p0[0], t1 = p0[1], t2 = p1[0], t3 = p1[1];
#pragma unroll
      for (int e = 0; e < 4; ++e) {
        afrag[f][e]      = (_Float16)t0[e];
        afrag[f][4 + e]  = (_Float16)t1[e];
        afrag[f][8 + e]  = (_Float16)t2[e];
        afrag[f][12 + e] = (_Float16)t3[e];
      }
    }
  }

  // ---------------- Phase A: scores = (Q K^T) * inv_temper ----------------
  for (int kt = wv; kt < S_ / 16; kt += 8) {
    const int n0 = kt * 16;
    v16h bf[4];          // preload all 4 B-frags -> one load clause, overlap
#pragma unroll
    for (int f = 0; f < 4; ++f) {
      const int kb = f * 32;
      if (PREP) {
        const _Float16* krow =
            kh + ((size_t)b * S_ + n0 + m) * D_ + kb + 16 * h;
        v8h t0 = *(const v8h*)krow;
        v8h t1 = *(const v8h*)(krow + 8);
#pragma unroll
        for (int e = 0; e < 8; ++e) { bf[f][e] = t0[e]; bf[f][8 + e] = t1[e]; }
      } else {
        const float* krow =
            k + ((size_t)b * S_ + n0 + m) * D_ + kb + 16 * h;
        const v4f* p = (const v4f*)krow;
        v4f t0 = p[0], t1 = p[1], t2 = p[2], t3 = p[3];
#pragma unroll
        for (int e = 0; e < 4; ++e) {
          bf[f][e]      = (_Float16)t0[e];
          bf[f][4 + e]  = (_Float16)t1[e];
          bf[f][8 + e]  = (_Float16)t2[e];
          bf[f][12 + e] = (_Float16)t3[e];
        }
      }
    }
    v8f c = {};
#pragma unroll
    for (int f = 0; f < 4; ++f)
      c = __builtin_amdgcn_wmma_f32_16x16x32_f16(false, afrag[f], false, bf[f],
                                                 (short)0, c, false, false);
#pragma unroll
    for (int g = 0; g < 8; ++g) {
      const int row = g + 8 * h;          // f32 C layout: VGPR g -> row g+8h
      sc[row * SSTR + n0 + m] = (_Float16)(c[g] * INV_TEMPER);
    }
  }
  __syncthreads();

  // ------- Phase B: softmax rows; write att; probs stay unnormalized -----
#pragma unroll
  for (int rr = 0; rr < 2; ++rr) {
    const int r = wv * 2 + rr;
    _Float16* srow = sc + r * SSTR;
    const int base = lane * 8;            // 8 halves per lane per step
    v8h sch[8];
#pragma unroll
    for (int j = 0; j < 8; ++j) sch[j] = *(const v8h*)(srow + base + j * 256);
    float mx = -3.4e38f;
#pragma unroll
    for (int j = 0; j < 8; ++j)
#pragma unroll
      for (int e = 0; e < 8; ++e) mx = fmaxf(mx, (float)sch[j][e]);
    mx = wave_max(mx);
    float sum = 0.f;
    v8h pch[8];
#pragma unroll
    for (int j = 0; j < 8; ++j) {
#pragma unroll
      for (int e = 0; e < 8; ++e) {
        float p = __expf((float)sch[j][e] - mx);
        pch[j][e] = (_Float16)p;
        sum += p;
      }
      *(v8h*)(srow + base + j * 256) = pch[j];  // unnormalized probs for C
    }
    sum = wave_sum(sum);
    const float rs = 1.f / sum;
    if (lane == 0) rs16[r] = rs;          // per-row scale for phase C epilogue
    float* arow = att + ((size_t)b * S_ + q0 + r) * S_;
#pragma unroll
    for (int j = 0; j < 8; ++j) {
      v4f o0, o1;
#pragma unroll
      for (int e = 0; e < 4; ++e) {
        o0[e] = (float)pch[j][e] * rs;
        o1[e] = (float)pch[j][4 + e] * rs;
      }
      *(v4f*)(arow + base + j * 256)     = o0;  // global_store_b128 x2
      *(v4f*)(arow + base + j * 256 + 4) = o1;
    }
  }
  __syncthreads();

  // ---------------- Phase C: out = (p * V) * rs[row] ----------------------
  const int d0 = wv * 16;                 // each wave: one 16-col slab of D
  v8f c = {};
  for (int kc = 0; kc < S_; kc += 32) {
    v16h a;                               // probs A-frag 16x32 from LDS
    {
      const _Float16* pr = sc + m * SSTR + kc + 16 * h;
      v8h t0 = *(const v8h*)pr;
      v8h t1 = *(const v8h*)(pr + 8);
#pragma unroll
      for (int e = 0; e < 8; ++e) { a[e] = t0[e]; a[8 + e] = t1[e]; }
    }
    v16h bfrag;                           // V B-frag: col n = d0+m, K = 16h+e
    if (PREP) {
      const _Float16* vrow =
          vt + ((size_t)b * D_ + d0 + m) * S_ + kc + 16 * h;
      v8h t0 = *(const v8h*)vrow;
      v8h t1 = *(const v8h*)(vrow + 8);
#pragma unroll
      for (int e = 0; e < 8; ++e) { bfrag[e] = t0[e]; bfrag[8 + e] = t1[e]; }
    } else {
      const float* vcol = v + ((size_t)b * S_ + kc + 16 * h) * D_ + d0 + m;
#pragma unroll
      for (int e = 0; e < 16; ++e) bfrag[e] = (_Float16)vcol[e * D_];
    }
    c = __builtin_amdgcn_wmma_f32_16x16x32_f16(false, a, false, bfrag,
                                               (short)0, c, false, false);
  }
#pragma unroll
  for (int g = 0; g < 8; ++g) {
    const int row = g + 8 * h;
    outp[((size_t)b * S_ + q0 + row) * D_ + d0 + m] = c[g] * rs16[row];
  }
}

extern "C" void kernel_launch(void* const* d_in, const int* in_sizes, int n_in,
                              void* d_out, int out_size, void* d_ws,
                              size_t ws_size, hipStream_t stream) {
  const float* q = (const float*)d_in[0];
  const float* k = (const float*)d_in[1];
  const float* v = (const float*)d_in[2];
  float* outp = (float*)d_out;
  float* att  = outp + (size_t)B_ * S_ * D_;   // tuple order: (out, att)

  const size_t elems = (size_t)B_ * S_ * D_;           // 16.8M
  const size_t need  = 2 * elems * sizeof(_Float16);   // Kh + Vt = 64 MB
  const int    nblk  = B_ * (S_ / QT);                 // 8192
  const size_t lds   = (size_t)QT * SSTR * sizeof(_Float16) + QT * sizeof(float);

  if (ws_size >= need) {
    _Float16* kh = (_Float16*)d_ws;
    _Float16* vt = kh + elems;
    cvt_f32_to_f16<<<(int)(elems / 4 / 256), 256, 0, stream>>>(
        k, kh, (int)(elems / 4));
    transpose_v<<<dim3(D_ / 32, S_ / 32, B_), dim3(32, 8), 0, stream>>>(v, vt);
    attn_kernel<true><<<nblk, 256, lds, stream>>>(q, k, v, kh, vt, outp, att);
  } else {
    attn_kernel<false><<<nblk, 256, lds, stream>>>(q, k, v, nullptr, nullptr,
                                                   outp, att);
  }
}
